// GAT_87720412054016
// MI455X (gfx1250) — matrix operand order repaired and checked by reading, not compiled, not run on hardware
//
#include <hip/hip_runtime.h>
#include <math.h>
#include <stdint.h>

// ---------------------------------------------------------------------------
// GAT (2-layer, DGL-style) for MI455X / gfx1250.
//  - Linear layers: v_wmma_f32_16x16x32_f16; W^T staged into LDS via the
//    Tensor Data Mover (tensor_load_to_lds + s_wait_tensorcnt) plus a
//    cooperative b128 copy (defines Bs for the compiler -> real ds ops).
//  - Edge softmax/scatter: u32-ordered atomicMax + f32 atomicAdd, with
//    global_prefetch of the next gather row.
// ---------------------------------------------------------------------------

typedef _Float16 v16h  __attribute__((ext_vector_type(16)));
typedef _Float16 v8h   __attribute__((ext_vector_type(8)));
typedef float    v8f   __attribute__((ext_vector_type(8)));
typedef unsigned u32x4 __attribute__((ext_vector_type(4)));
typedef int      i32x8 __attribute__((ext_vector_type(8)));

#define CH 128          // channels per node in both layers (H*F == 128)

// ---------------- fp32 -> fp16 conversion ----------------------------------
__global__ void f32_to_f16_k(const float* __restrict__ in,
                             _Float16* __restrict__ out, int n) {
  int i = blockIdx.x * blockDim.x + threadIdx.x;
  if (i < n) out[i] = (_Float16)in[i];
}

// ---------------- fp32 -> fp16 with transpose (weights) --------------------
// WT[n][k] = (f16) W[k][n], so B fragments become contiguous per lane.
__global__ void w_to_f16T_k(const float* __restrict__ W,
                            _Float16* __restrict__ WT) {
  int i = blockIdx.x * blockDim.x + threadIdx.x;   // over CH*CH, coalesced read
  if (i >= CH * CH) return;
  int k = i / CH, n = i % CH;
  WT[n * CH + k] = (_Float16)W[i];
}

// ---------------- WMMA GEMM: C[M x 128] = A[M x 128] * W[128 x 128] --------
// BT = W^T (fp16, row-major [n][k]).  One wave per 16x16 C tile.
// W^T (32 KB) is staged into LDS (TDM + cooperative copy); K is consumed by
// 4 x v_wmma_f32_16x16x32_f16 on two independent accumulator chains.
__global__ __launch_bounds__(128)
void gemm_wmma_k(const _Float16* __restrict__ A, const _Float16* __restrict__ BT,
                 float* __restrict__ C, int M) {
  __shared__ _Float16 Bs[CH * CH];                 // 32 KB, only LDS object -> offset 0

  // ---- TDM: 2-D tile (128 x 128, 2-byte elems) global -> LDS[0] ----
  if (threadIdx.x == 0) {
    unsigned long long ga = (unsigned long long)(uintptr_t)BT;
    u32x4 g0;
    g0.x = 1u;                                      // count = 1 (valid D#)
    g0.y = 0u;                                      // lds_addr = 0
    g0.z = (unsigned)(ga & 0xFFFFFFFFu);            // global_addr[31:0]  (bits 95:64)
    g0.w = (unsigned)((ga >> 32) & 0x01FFFFFFu)     // global_addr[56:32] (bits 120:96)
         | (2u << 30);                              // type = 2 ("image")
    i32x8 g1;
    g1[0] = 0x00010000;                             // data_size = 1 -> 2-byte elems
    g1[1] = (int)(CH << 16);                        // tensor_dim0 = 128 (bits 79:48)
    g1[2] = (int)(CH << 16);                        // tensor_dim1 = 128 (bits 111:80)
    g1[3] = (int)(CH << 16);                        // tile_dim0   = 128 (bits 127:112)
    g1[4] = CH;                                     // tile_dim1   = 128 (bits 143:128)
    g1[5] = CH;                                     // tensor_dim0_stride = 128 (bits 207:160)
    g1[6] = 0;
    g1[7] = 0;
    asm volatile("tensor_load_to_lds %0, %1" :: "s"(g0), "s"(g1) : "memory");
    __builtin_amdgcn_s_wait_tensorcnt(0);
  }
  // ---- cooperative stage (defines Bs; global_load_b128 -> ds_store_b128) --
  {
    v8h* __restrict__       bs = (v8h*)Bs;
    const v8h* __restrict__ bt = (const v8h*)BT;
#pragma unroll
    for (int i = 0; i < (CH * CH / 8) / 128; ++i)   // 16 chunks of 16B / thread
      bs[threadIdx.x + i * 128] = bt[threadIdx.x + i * 128];
  }
  __syncthreads();

  int wave = threadIdx.x >> 5;
  int lane = threadIdx.x & 31;
  int tile = blockIdx.x * 4 + wave;
  int row0 = (tile >> 3) * 16;                     // CH/16 == 8 column tiles
  int col0 = (tile & 7) * 16;
  int  lm  = lane & 15;
  bool hi  = lane >= 16;

  if (row0 < M) {
    v8f c0 = {0.f, 0.f, 0.f, 0.f, 0.f, 0.f, 0.f, 0.f};
    v8f c1 = {0.f, 0.f, 0.f, 0.f, 0.f, 0.f, 0.f, 0.f};
    // A fragment: halves 0-7 at K = off+0..7, halves 8-15 at K = off+16..23
    const _Float16* Arow = A + (size_t)(row0 + lm) * CH + (hi ? 8 : 0);
    // B fragment (from W^T in LDS): halves 0-7 at K = off+0..7, halves 8-15 at K = off+16..23
    const _Float16* Brow = Bs + (size_t)(col0 + lm) * CH + (hi ? 8 : 0);
#pragma unroll
    for (int k0 = 0; k0 < CH; k0 += 64) {          // 2 WMMAs / iter, 2 chains
      v8h a0lo = *(const v8h*)(Arow + k0);
      v8h a0hi = *(const v8h*)(Arow + k0 + 16);
      v8h a1lo = *(const v8h*)(Arow + k0 + 32);
      v8h a1hi = *(const v8h*)(Arow + k0 + 48);
      v16h a0 = __builtin_shufflevector(a0lo, a0hi, 0, 1, 2, 3, 4, 5, 6, 7,
                                                    8, 9, 10, 11, 12, 13, 14, 15);
      v16h a1 = __builtin_shufflevector(a1lo, a1hi, 0, 1, 2, 3, 4, 5, 6, 7,
                                                    8, 9, 10, 11, 12, 13, 14, 15);
      v16h b0 = __builtin_shufflevector(*(const v8h*)(Brow + k0), *(const v8h*)(Brow + k0 + 16), 0, 1, 2, 3, 4, 5, 6, 7, 8, 9, 10, 11, 12, 13, 14, 15);         // 2 x ds_load_b128
      v16h b1 = __builtin_shufflevector(*(const v8h*)(Brow + k0 + 32), *(const v8h*)(Brow + k0 + 48), 0, 1, 2, 3, 4, 5, 6, 7, 8, 9, 10, 11, 12, 13, 14, 15);    // 2 x ds_load_b128
      c0 = __builtin_amdgcn_wmma_f32_16x16x32_f16(false, a0, false, b0,
                                                  (short)0, c0, false, false);
      c1 = __builtin_amdgcn_wmma_f32_16x16x32_f16(false, a1, false, b1,
                                                  (short)0, c1, false, false);
    }
    v8f c = c0 + c1;
    // C/D layout: VGPR r -> row r (lanes 0-15) / row r+8 (lanes 16-31)
    float* Crow = C + (size_t)(row0 + (hi ? 8 : 0)) * CH + col0 + lm;
#pragma unroll
    for (int r = 0; r < 8; ++r) Crow[(size_t)r * CH] = c[r];
  }
}

// ---------------- per-node attention scores --------------------------------
__global__ void node_scores_k(const float* __restrict__ feat,
                              const float* __restrict__ attn_l,
                              const float* __restrict__ attn_r,
                              float* __restrict__ el, float* __restrict__ er,
                              int N, int H, int F) {
  int i = blockIdx.x * blockDim.x + threadIdx.x;   // over N*H
  if (i >= N * H) return;
  int n = i / H, h = i % H;
  const float* fp = feat + (size_t)n * CH + h * F;
  const float* al = attn_l + h * F;
  const float* ar = attn_r + h * F;
  float sl = 0.f, sr = 0.f;
  for (int f = 0; f < F; ++f) { float v = fp[f]; sl += v * al[f]; sr += v * ar[f]; }
  el[i] = sl; er[i] = sr;
}

// ---------------- order-preserving float<->uint for atomicMax --------------
__device__ __forceinline__ unsigned f32_ord(float f) {
  unsigned u = __float_as_uint(f);
  return (u & 0x80000000u) ? ~u : (u | 0x80000000u);
}
__device__ __forceinline__ float ord_f32(unsigned u) {
  return (u & 0x80000000u) ? __uint_as_float(u & 0x7FFFFFFFu)
                           : __uint_as_float(~u);
}
__device__ __forceinline__ float leaky(float x) { return x > 0.f ? x : 0.2f * x; }

// ---------------- edge pass 1: segment max of leaky_relu scores ------------
__global__ void edge_max_k(const int* __restrict__ src, const int* __restrict__ dst,
                           const float* __restrict__ el, const float* __restrict__ er,
                           unsigned* __restrict__ emax, int E, int H) {
  int i = blockIdx.x * blockDim.x + threadIdx.x;   // over E*H
  if (i >= E * H) return;
  int e = i / H, h = i % H;
  int s = src[e], d = dst[e];
  float sc = leaky(el[s * H + h] + er[d * H + h]);
  atomicMax(&emax[d * H + h], f32_ord(sc));
}

// ---------------- edge pass 2: segment sum of exp --------------------------
__global__ void edge_denom_k(const int* __restrict__ src, const int* __restrict__ dst,
                             const float* __restrict__ el, const float* __restrict__ er,
                             const unsigned* __restrict__ emax,
                             float* __restrict__ den, int E, int H) {
  int i = blockIdx.x * blockDim.x + threadIdx.x;
  if (i >= E * H) return;
  int e = i / H, h = i % H;
  int s = src[e], d = dst[e];
  int idx = d * H + h;
  float sc = leaky(el[s * H + h] + er[idx]);
  atomicAdd(&den[idx], expf(sc - ord_f32(emax[idx])));
}

// ---------------- edge pass 3: weighted scatter-sum of messages ------------
// Grid-stride over edges, 128 threads per edge; alpha computed in lane 0 and
// broadcast (h = tid/F is wave-uniform since F >= 32).  Prefetch next gather.
__global__ __launch_bounds__(128)
void edge_msg_k(const int* __restrict__ src, const int* __restrict__ dst,
                const float* __restrict__ el, const float* __restrict__ er,
                const unsigned* __restrict__ emax, const float* __restrict__ den,
                const float* __restrict__ feat, float* __restrict__ rst,
                int E, int H, int F) {
  int i = threadIdx.x;                 // 0..127 == h*F + f
  int h = i / F;
  int lane = threadIdx.x & 31;
  for (int e = blockIdx.x; e < E; e += gridDim.x) {
    int s = src[e], d = dst[e];
    int en = e + gridDim.x;
    if (en < E)                        // global_prefetch_b8 of next source row
      __builtin_prefetch(&feat[(size_t)src[en] * CH + i], 0, 0);
    float alpha = 0.f;
    if (lane == 0) {
      int idx = d * H + h;
      float sc = leaky(el[s * H + h] + er[idx]);
      alpha = expf(sc - ord_f32(emax[idx])) / den[idx];
    }
    alpha = __shfl(alpha, 0, 32);
    atomicAdd(&rst[(size_t)d * CH + i], feat[(size_t)s * CH + i] * alpha);
  }
}

// ---------------- finalize: bias + ELU (+ optional fp16 emit) --------------
__global__ void finalize_k(const float* __restrict__ rst, const float* __restrict__ bias,
                           float* __restrict__ out32, _Float16* __restrict__ out16,
                           int total) {
  int i = blockIdx.x * blockDim.x + threadIdx.x;
  if (i >= total) return;
  float v = rst[i] + bias[i & (CH - 1)];
  v = v > 0.f ? v : expm1f(v);
  if (out32) out32[i] = v;
  if (out16) out16[i] = (_Float16)v;
}

// ---------------------------------------------------------------------------
extern "C" void kernel_launch(void* const* d_in, const int* in_sizes, int n_in,
                              void* d_out, int out_size, void* d_ws, size_t ws_size,
                              hipStream_t stream) {
  const float* x       = (const float*)d_in[0];
  const int*   src     = (const int*)d_in[1];
  const int*   dst     = (const int*)d_in[2];
  const float* W0      = (const float*)d_in[3];
  const float* attn_l0 = (const float*)d_in[4];
  const float* attn_r0 = (const float*)d_in[5];
  const float* bias0   = (const float*)d_in[6];
  const float* W1      = (const float*)d_in[7];
  const float* attn_l1 = (const float*)d_in[8];
  const float* attn_r1 = (const float*)d_in[9];
  const float* bias1   = (const float*)d_in[10];

  const int N  = in_sizes[0] / CH;
  const int E  = in_sizes[1];
  const int H0 = 4, F0 = 32;
  const int nNC = N * CH;

  // workspace carve-out (256B aligned)
  char* ws = (char*)d_ws;
  size_t off = 0;
  auto carve = [&](size_t bytes) -> void* {
    void* p = ws + off;
    off = (off + bytes + 255) & ~(size_t)255;
    return p;
  };
  _Float16* x_h  = (_Float16*)carve((size_t)nNC * sizeof(_Float16)); // reused as h fp16
  _Float16* w_h  = (_Float16*)carve((size_t)CH * CH * sizeof(_Float16)); // W^T fp16
  float*    feat = (float*)carve((size_t)nNC * sizeof(float));
  float*    el   = (float*)carve((size_t)N * H0 * sizeof(float));
  float*    er   = (float*)carve((size_t)N * H0 * sizeof(float));
  unsigned* emax = (unsigned*)carve((size_t)N * H0 * sizeof(unsigned));
  float*    den  = (float*)carve((size_t)N * H0 * sizeof(float));
  float*    rst  = (float*)carve((size_t)nNC * sizeof(float));
  float*    out  = (float*)d_out;

  const int T = 256;
  const int gemmBlocks = ((N / 16) * (CH / 16) + 3) / 4;  // 4 waves/block
  const int msgBlocks  = 8192;                            // grid-stride edges

  // ======================= Layer 0: GATConv(128 -> 32, 4 heads) ============
  hipMemsetAsync(emax, 0, (size_t)N * H0 * sizeof(unsigned), stream);
  hipMemsetAsync(den,  0, (size_t)N * H0 * sizeof(float), stream);
  hipMemsetAsync(rst,  0, (size_t)nNC * sizeof(float), stream);
  hipMemsetAsync(out,  0, (size_t)nNC * sizeof(float), stream);

  f32_to_f16_k<<<(nNC + T - 1) / T, T, 0, stream>>>(x, x_h, nNC);
  w_to_f16T_k<<<(CH * CH + T - 1) / T, T, 0, stream>>>(W0, w_h);
  gemm_wmma_k<<<gemmBlocks, 128, 0, stream>>>(x_h, w_h, feat, N);
  node_scores_k<<<(N * H0 + T - 1) / T, T, 0, stream>>>(feat, attn_l0, attn_r0,
                                                        el, er, N, H0, F0);
  edge_max_k<<<(E * H0 + T - 1) / T, T, 0, stream>>>(src, dst, el, er, emax, E, H0);
  edge_denom_k<<<(E * H0 + T - 1) / T, T, 0, stream>>>(src, dst, el, er, emax, den, E, H0);
  edge_msg_k<<<msgBlocks, 128, 0, stream>>>(src, dst, el, er, emax, den, feat,
                                            rst, E, H0, F0);
  finalize_k<<<(nNC + T - 1) / T, T, 0, stream>>>(rst, bias0, nullptr, x_h, nNC);

  // ======================= Layer 1: GATConv(128 -> 128, 1 head) ============
  hipMemsetAsync(emax, 0, (size_t)N * sizeof(unsigned), stream);
  hipMemsetAsync(den,  0, (size_t)N * sizeof(float), stream);

  w_to_f16T_k<<<(CH * CH + T - 1) / T, T, 0, stream>>>(W1, w_h);
  gemm_wmma_k<<<gemmBlocks, 128, 0, stream>>>(x_h, w_h, feat, N);
  node_scores_k<<<(N + T - 1) / T, T, 0, stream>>>(feat, attn_l1, attn_r1,
                                                   el, er, N, 1, CH);
  edge_max_k<<<(E + T - 1) / T, T, 0, stream>>>(src, dst, el, er, emax, E, 1);
  edge_denom_k<<<(E + T - 1) / T, T, 0, stream>>>(src, dst, el, er, emax, den, E, 1);
  edge_msg_k<<<msgBlocks, 128, 0, stream>>>(src, dst, el, er, emax, den, feat,
                                            out, E, 1, CH);
  finalize_k<<<(nNC + T - 1) / T, T, 0, stream>>>(out, bias1, out, nullptr, nNC);
  // mean over heads == identity (1 head) -> d_out holds [N, 128] fp32.
}
